// HPNF_RST_82506321756622
// MI455X (gfx1250) — matrix operand order repaired and checked by reading, not compiled, not run on hardware
//
#include <hip/hip_runtime.h>
#include <hip/hip_bf16.h>

// ---------------------------------------------------------------------------
// HPNF_RST: 2-layer GCN + BN + pool + MLP head for MI455X (gfx1250, wave32).
// Dense GEMMs (100000x128 @ 128x128) use v_wmma_f32_16x16x32_f16 with
// ds_load_b128 fragment loads (LDS layouts match the ISA VGPR striping).
// Edge scatter (the HBM-bound part) uses one wave per edge, float4 gather,
// global_atomic_add_f32 scatter.
// ---------------------------------------------------------------------------

typedef __attribute__((ext_vector_type(16))) _Float16 v16h;
typedef __attribute__((ext_vector_type(8)))  _Float16 v8h;
typedef __attribute__((ext_vector_type(4)))  _Float16 v4h;
typedef __attribute__((ext_vector_type(8)))  float    v8f;

#define KDIM 128   // input channels of both convs
#define NCOL 128   // output channels of both convs
#define ROWS_PER_BLOCK 128
#define LDS_PITCH 136   // 128 + 8 halves pad: 272B row stride (16B-aligned, bank-skewed)
#define BN_EPS 1e-5f

// ---------------- degree / norm ----------------
__global__ void k_fill_deg(float* __restrict__ deg, int n) {
    int i = blockIdx.x * blockDim.x + threadIdx.x;
    if (i < n) deg[i] = 1.0f;                       // self-loop
}

__global__ void k_deg_edges(const int* __restrict__ dst, float* __restrict__ deg, int e) {
    int i = blockIdx.x * blockDim.x + threadIdx.x;
    if (i < e) atomicAdd(&deg[dst[i]], 1.0f);
}

__global__ void k_dinv(float* __restrict__ deg, int n) {
    int i = blockIdx.x * blockDim.x + threadIdx.x;
    if (i < n) deg[i] = rsqrtf(deg[i]);             // deg >= 1 always
}

// ---------------- WMMA GEMM: Y[n x 128] = act(X)[n x 128] @ W[128 x 128] ---
// If bnscale != nullptr, loader applies per-channel x*scale+shift then ReLU.
__global__ __launch_bounds__(256)
void k_gemm_wmma(const float* __restrict__ X, const float* __restrict__ W,
                 float* __restrict__ Y, int n,
                 const float* __restrict__ bnscale, const float* __restrict__ bnshift)
{
    __shared__ _Float16 sWt[NCOL][LDS_PITCH];           // W transposed: [ncol][k]
    __shared__ _Float16 sX [ROWS_PER_BLOCK][LDS_PITCH]; // [row][k]

    const int tid   = threadIdx.x;
    const int wave  = tid >> 5;
    const int lane  = tid & 31;
    const int rbase = blockIdx.x * ROWS_PER_BLOCK;

    // Stage W transposed (f32 -> f16): sWt[n][k] = W[k][n].
    // Each thread packs 4 consecutive k for one n -> one 8B ds_store_b64.
    for (int i = tid; i < (KDIM / 4) * NCOL; i += 256) {
        int k4 = (i >> 7) * 4;      // 0,4,...,124
        int nn = i & 127;           // consecutive lanes -> consecutive n (coalesced)
        v4h pk;
        pk[0] = (_Float16)W[(k4 + 0) * NCOL + nn];
        pk[1] = (_Float16)W[(k4 + 1) * NCOL + nn];
        pk[2] = (_Float16)W[(k4 + 2) * NCOL + nn];
        pk[3] = (_Float16)W[(k4 + 3) * NCOL + nn];
        *(v4h*)&sWt[nn][k4] = pk;
    }
    // Stage X tile (f32 -> f16), optional fused BN + ReLU, zero-pad tail rows.
    for (int i = tid; i < (ROWS_PER_BLOCK * KDIM) / 4; i += 256) {
        int r  = i >> 5;            // 32 float4 per row
        int k4 = (i & 31) * 4;
        int grow = rbase + r;
        float4 v = make_float4(0.f, 0.f, 0.f, 0.f);
        if (grow < n) v = ((const float4*)(X + (size_t)grow * KDIM))[i & 31];
        if (bnscale) {
            v.x = fmaxf(v.x * bnscale[k4 + 0] + bnshift[k4 + 0], 0.f);
            v.y = fmaxf(v.y * bnscale[k4 + 1] + bnshift[k4 + 1], 0.f);
            v.z = fmaxf(v.z * bnscale[k4 + 2] + bnshift[k4 + 2], 0.f);
            v.w = fmaxf(v.w * bnscale[k4 + 3] + bnshift[k4 + 3], 0.f);
        }
        v4h pk;
        pk[0] = (_Float16)v.x; pk[1] = (_Float16)v.y;
        pk[2] = (_Float16)v.z; pk[3] = (_Float16)v.w;
        *(v4h*)&sX[r][k4] = pk;
    }
    __syncthreads();

    // Each wave computes rows [wave*16, wave*16+16) x all 128 cols (8 tiles)
    v8f acc[8] = {};
    const int mrow  = wave * 16 + (lane & 15);      // A row (M) for this lane
    const int aKhi  = (lane >= 16) ? 8 : 0;         // A K-offset per ISA layout
    const int bKhi  = (lane >= 16) ? 8 : 0;         // B K-offset per ISA layout
    const int ncol0 = lane & 15;                    // B/C column for this lane

    #pragma unroll
    for (int kc = 0; kc < KDIM; kc += 32) {
        // A fragment: halves k = kc+aKhi+{0..7} and kc+aKhi+{16..23}
        //  -> two contiguous 16B runs -> 2x ds_load_b128
        v8h alo = *(const v8h*)&sX[mrow][kc + aKhi];
        v8h ahi = *(const v8h*)&sX[mrow][kc + aKhi + 16];
        v16h a = __builtin_shufflevector(alo, ahi,
                  0, 1, 2, 3, 4, 5, 6, 7, 8, 9, 10, 11, 12, 13, 14, 15);
        #pragma unroll
        for (int nt = 0; nt < 8; ++nt) {
            // B fragment: column nt*16+ncol0, k = kc+bKhi+{0..7} and kc+bKhi+{16..23}
            //  -> two contiguous 16B runs -> 2x ds_load_b128
            const _Float16* bp = &sWt[nt * 16 + ncol0][kc + bKhi];
            v8h blo = *(const v8h*)(bp);
            v8h bhi = *(const v8h*)(bp + 16);
            v16h b = __builtin_shufflevector(blo, bhi,
                      0, 1, 2, 3, 4, 5, 6, 7, 8, 9, 10, 11, 12, 13, 14, 15);
            acc[nt] = __builtin_amdgcn_wmma_f32_16x16x32_f16(
                false, a, false, b, (short)0, acc[nt], false, false);
        }
    }

    // Store: C/D layout — VGPR r: lanes 0-15 -> M=r, lanes 16-31 -> M=r+8
    const int rlocal = wave * 16 + ((lane >= 16) ? 8 : 0);
    #pragma unroll
    for (int nt = 0; nt < 8; ++nt) {
        #pragma unroll
        for (int r = 0; r < 8; ++r) {
            int grow = rbase + rlocal + r;
            if (grow < n)
                Y[(size_t)grow * NCOL + nt * 16 + ncol0] = acc[nt][r];
        }
    }
}

// ---------------- scatter: init with bias + self-loop term ----------------
__global__ void k_scatter_init(const float* __restrict__ h, const float* __restrict__ dinv,
                               const float* __restrict__ bias, float* __restrict__ agg, int n)
{
    int gid  = blockIdx.x * blockDim.x + threadIdx.x;
    int node = gid >> 5;
    int lane = gid & 31;
    if (node >= n) return;
    float di = dinv[node];
    float nrm = di * di;                            // self-loop norm
    int c = lane * 4;
    const float4 v = *(const float4*)(h + (size_t)node * NCOL + c);
    float4 o;
    o.x = bias[c + 0] + v.x * nrm;
    o.y = bias[c + 1] + v.y * nrm;
    o.z = bias[c + 2] + v.z * nrm;
    o.w = bias[c + 3] + v.w * nrm;
    *(float4*)(agg + (size_t)node * NCOL + c) = o;
}

// ---------------- scatter: per-edge atomic accumulate ----------------------
__global__ void k_scatter_edges(const float* __restrict__ h, const int* __restrict__ src,
                                const int* __restrict__ dst, const float* __restrict__ dinv,
                                float* __restrict__ agg, int e)
{
    int gid  = blockIdx.x * blockDim.x + threadIdx.x;
    int edge = gid >> 5;                            // one wave32 per edge
    int lane = gid & 31;
    if (edge >= e) return;
    int s = src[edge], d = dst[edge];
    float nrm = dinv[s] * dinv[d];
    int c = lane * 4;
    const float4 v = *(const float4*)(h + (size_t)s * NCOL + c);
    float* o = agg + (size_t)d * NCOL + c;
    atomicAdd(o + 0, v.x * nrm);
    atomicAdd(o + 1, v.y * nrm);
    atomicAdd(o + 2, v.z * nrm);
    atomicAdd(o + 3, v.w * nrm);
}

// ---------------- batch-norm statistics ------------------------------------
#define BN_ROWS 256
__global__ __launch_bounds__(128)
void k_bn_stats(const float* __restrict__ h, float* __restrict__ stats, int n)
{
    int c  = threadIdx.x;                           // channel, 128 threads
    int r0 = blockIdx.x * BN_ROWS;
    int r1 = min(r0 + BN_ROWS, n);
    float s = 0.f, q = 0.f;
    for (int r = r0; r < r1; ++r) {
        float v = h[(size_t)r * NCOL + c];
        s += v; q += v * v;
    }
    atomicAdd(&stats[c], s);
    atomicAdd(&stats[NCOL + c], q);
}

__global__ void k_bn_final(const float* __restrict__ stats,
                           const float* __restrict__ gamma, const float* __restrict__ beta,
                           float* __restrict__ scale, float* __restrict__ shift, int n)
{
    int c = threadIdx.x;
    if (c >= NCOL) return;
    float inv_n = 1.0f / (float)n;
    float mu  = stats[c] * inv_n;
    float var = stats[NCOL + c] * inv_n - mu * mu;
    float istd = rsqrtf(var + BN_EPS);
    float sc = gamma[c] * istd;
    scale[c] = sc;
    shift[c] = beta[c] - mu * sc;
}

// ---------------- pooling --------------------------------------------------
__global__ void k_pool(const float* __restrict__ h, const int* __restrict__ batch,
                       float* __restrict__ sums, float* __restrict__ counts, int n)
{
    int gid  = blockIdx.x * blockDim.x + threadIdx.x;
    int node = gid >> 5;
    int lane = gid & 31;
    if (node >= n) return;
    int g = batch[node];
    int c = lane * 4;
    const float4 v = *(const float4*)(h + (size_t)node * NCOL + c);
    float* o = sums + (size_t)g * NCOL + c;
    atomicAdd(o + 0, v.x);
    atomicAdd(o + 1, v.y);
    atomicAdd(o + 2, v.z);
    atomicAdd(o + 3, v.w);
    if (lane == 0) atomicAdd(&counts[g], 1.0f);
}

__global__ void k_pool_div(float* __restrict__ sums, const float* __restrict__ counts, int G)
{
    int gid = blockIdx.x * blockDim.x + threadIdx.x;
    if (gid >= G * NCOL) return;
    int g = gid / NCOL;
    sums[gid] /= fmaxf(counts[g], 1.0f);
}

// ---------------- small head GEMMs -----------------------------------------
__global__ void k_lin_relu(const float* __restrict__ X, const float* __restrict__ W,
                           const float* __restrict__ b, float* __restrict__ Y,
                           int G, int Kin, int Nout)
{
    int gid = blockIdx.x * blockDim.x + threadIdx.x;
    if (gid >= G * Nout) return;
    int g = gid / Nout, o = gid % Nout;
    float acc = b[o];
    for (int k = 0; k < Kin; ++k) acc += X[g * Kin + k] * W[k * Nout + o];
    Y[gid] = fmaxf(acc, 0.f);
}

__global__ void k_head_out(const float* __restrict__ xg, const float* __restrict__ xr,
                           const float* __restrict__ Wc, const float* __restrict__ bc,
                           float* __restrict__ out, int G)
{
    int gid = blockIdx.x * blockDim.x + threadIdx.x;
    if (gid >= G * 2) return;
    int g = gid / 2, o = gid % 2;
    float acc = bc[o];
    for (int k = 0; k < 64; ++k) acc += xg[g * 64 + k] * Wc[k * 2 + o];
    for (int k = 0; k < 64; ++k) acc += xr[g * 64 + k] * Wc[(64 + k) * 2 + o];
    out[gid] = acc;
}

// ---------------------------------------------------------------------------
extern "C" void kernel_launch(void* const* d_in, const int* in_sizes, int n_in,
                              void* d_out, int out_size, void* d_ws, size_t ws_size,
                              hipStream_t stream)
{
    // setup_inputs() order:
    // 0:x 1:edge_index 2:batch 3:rst 4:num_graphs 5:W1 6:b1 7:gamma 8:beta
    // 9:W2 10:b2 11:Wg 12:bg 13:Wr 14:br 15:Wc 16:bc
    const float* x     = (const float*)d_in[0];
    const int*   ei    = (const int*)  d_in[1];
    const int*   batch = (const int*)  d_in[2];
    const float* rst   = (const float*)d_in[3];
    const float* W1    = (const float*)d_in[5];
    const float* b1    = (const float*)d_in[6];
    const float* gamma = (const float*)d_in[7];
    const float* beta  = (const float*)d_in[8];
    const float* W2    = (const float*)d_in[9];
    const float* b2    = (const float*)d_in[10];
    const float* Wg    = (const float*)d_in[11];
    const float* bg    = (const float*)d_in[12];
    const float* Wr    = (const float*)d_in[13];
    const float* br    = (const float*)d_in[14];
    const float* Wc    = (const float*)d_in[15];
    const float* bc    = (const float*)d_in[16];
    float* out = (float*)d_out;

    const int n = in_sizes[0] / KDIM;       // 100000 nodes
    const int E = in_sizes[1] / 2;          // 1600000 edges
    const int G = in_sizes[3] / 64;         // 512 graphs (rst is [G,64])
    const int* src = ei;
    const int* dst = ei + E;

    // ---- workspace carve-up (256B aligned) ----
    char* ws = (char*)d_ws;
    size_t off = 0;
    auto alloc = [&](size_t bytes) -> void* {
        off = (off + 255) & ~(size_t)255;
        void* p = ws + off;
        off += bytes;
        return p;
    };
    float* dinv    = (float*)alloc((size_t)n * sizeof(float));
    float* bufA    = (float*)alloc((size_t)n * NCOL * sizeof(float)); // h1 then h2
    float* bufB    = (float*)alloc((size_t)n * NCOL * sizeof(float)); // agg1 then agg2
    float* stats   = (float*)alloc(2 * NCOL * sizeof(float));
    float* bnscale = (float*)alloc(NCOL * sizeof(float));
    float* bnshift = (float*)alloc(NCOL * sizeof(float));
    float* psums   = (float*)alloc((size_t)G * NCOL * sizeof(float));
    float* pcnts   = (float*)alloc((size_t)G * sizeof(float));
    float* xg      = (float*)alloc((size_t)G * 64 * sizeof(float));
    float* xr      = (float*)alloc((size_t)G * 64 * sizeof(float));
    (void)ws_size;

    const int T = 256;
    dim3 blk(T);
    int gN      = (n + T - 1) / T;
    int gE      = (E + T - 1) / T;
    int gNodeW  = (int)(((size_t)n * 32 + T - 1) / T);
    int gEdgeW  = (int)(((size_t)E * 32 + T - 1) / T);
    int gGemm   = (n + ROWS_PER_BLOCK - 1) / ROWS_PER_BLOCK;
    int gBN     = (n + BN_ROWS - 1) / BN_ROWS;

    // degree -> dinv
    k_fill_deg <<<gN, blk, 0, stream>>>(dinv, n);
    k_deg_edges<<<gE, blk, 0, stream>>>(dst, dinv, E);
    k_dinv     <<<gN, blk, 0, stream>>>(dinv, n);

    // conv1: h1 = x @ W1 (WMMA), then agg1 = b1 + self-loop + edge scatter
    k_gemm_wmma   <<<gGemm, blk, 0, stream>>>(x, W1, bufA, n, nullptr, nullptr);
    k_scatter_init<<<gNodeW, blk, 0, stream>>>(bufA, dinv, b1, bufB, n);
    k_scatter_edges<<<gEdgeW, blk, 0, stream>>>(bufA, src, dst, dinv, bufB, E);

    // batch-norm statistics over agg1
    hipMemsetAsync(stats, 0, 2 * NCOL * sizeof(float), stream);
    k_bn_stats<<<gBN, dim3(128), 0, stream>>>(bufB, stats, n);
    k_bn_final<<<1, dim3(128), 0, stream>>>(stats, gamma, beta, bnscale, bnshift, n);

    // conv2: h2 = relu(BN(agg1)) @ W2 (BN+ReLU fused into WMMA loader)
    k_gemm_wmma   <<<gGemm, blk, 0, stream>>>(bufB, W2, bufA, n, bnscale, bnshift);
    k_scatter_init<<<gNodeW, blk, 0, stream>>>(bufA, dinv, b2, bufB, n);
    k_scatter_edges<<<gEdgeW, blk, 0, stream>>>(bufA, src, dst, dinv, bufB, E);

    // global mean pool
    hipMemsetAsync(psums, 0, (size_t)G * NCOL * sizeof(float), stream);
    hipMemsetAsync(pcnts, 0, (size_t)G * sizeof(float), stream);
    k_pool    <<<gNodeW, blk, 0, stream>>>(bufB, batch, psums, pcnts, n);
    k_pool_div<<<(G * NCOL + T - 1) / T, blk, 0, stream>>>(psums, pcnts, G);

    // head
    k_lin_relu<<<(G * 64 + T - 1) / T, blk, 0, stream>>>(psums, Wg, bg, xg, G, 128, 64);
    k_lin_relu<<<(G * 64 + T - 1) / T, blk, 0, stream>>>(rst,   Wr, br, xr, G, 64, 64);
    k_head_out<<<(G * 2 + T - 1) / T, blk, 0, stream>>>(xg, xr, Wc, bc, out, G);
}
